// TransformerMemory_82918638617246
// MI455X (gfx1250) — matrix operand order repaired and checked by reading, not compiled, not run on hardware
//
#include <hip/hip_runtime.h>
#include <hip/hip_bf16.h>
#include <cstdint>

typedef __attribute__((ext_vector_type(16))) _Float16     v16h;
typedef __attribute__((ext_vector_type(8)))  float        v8f;
typedef __attribute__((ext_vector_type(4)))  unsigned int v4u;
typedef __attribute__((ext_vector_type(8)))  int          v8i;
typedef __attribute__((ext_vector_type(4)))  int          v4i;

#define SEQ   500
#define BATCH 32
#define DM    512
#define NH    8
#define HD    64
#define NL    2
#define FFD   2048
#define WIN   100
#define MROWS (SEQ*BATCH)   // 16000

#define BM 128
#define BN 128
#define BK 32
#define BKP 40   // 32 halves + 4-DWORD pad inserted by TDM per row

// ---------------- TDM tile load: f16 2D tile [tileH x 32] from [H x K] row-major ----
__device__ __forceinline__ void tdm_load_tile(unsigned lds_byte_off,
                                              const _Float16* tile_ptr,
                                              int K_, int H_, int tileH) {
#if __has_builtin(__builtin_amdgcn_tensor_load_to_lds)
  const unsigned long long ga = (unsigned long long)(uintptr_t)tile_ptr;
  v4u g0;
  g0[0] = 1u;                                        // count=1 (valid user D#)
  g0[1] = lds_byte_off;                              // lds_addr
  g0[2] = (unsigned)(ga & 0xFFFFFFFFu);              // global_addr[31:0]
  g0[3] = (unsigned)((ga >> 32) & 0x1FFFFFFu) | 0x80000000u;  // [56:32] | type=2
  v8i g1;
  // data_size=2B(1<<16), pad_enable(1<<20), pad_interval=16dw(3<<22), pad_amount=4dw(3<<25)
  g1[0] = (int)((1u << 16) | (1u << 20) | (3u << 22) | (3u << 25));
  g1[1] = (int)((unsigned)(K_ & 0xFFFF) << 16);                       // tensor_dim0 lo
  g1[2] = (int)(((unsigned)K_ >> 16) | ((unsigned)(H_ & 0xFFFF) << 16)); // dim0 hi | dim1 lo
  g1[3] = (int)(((unsigned)H_ >> 16) | ((unsigned)BK << 16));         // dim1 hi | tile_dim0=32
  g1[4] = tileH;                                                      // tile_dim1
  g1[5] = K_;                                                         // tensor_dim0_stride
  g1[6] = 0;
  g1[7] = 0;
  v4i gz = {};
#if defined(__clang_major__) && __clang_major__ >= 23
  v8i gz8 = {};
  __builtin_amdgcn_tensor_load_to_lds(g0, g1, gz, gz, gz8, 0);
#else
  __builtin_amdgcn_tensor_load_to_lds(g0, g1, gz, gz, 0);
#endif
#endif
}

// ---------------- WMMA GEMM with TDM double-buffered staging ----------------
// C[M,N] (+f16 copy) = epilogue(A16[M,K] @ Bt16[N,K]^T); 256 threads = 8 waves (2M x 4N),
// each wave: 4x2 tiles of 16x16 via v_wmma_f32_16x16x32_f16.
__global__ __launch_bounds__(256)
void gemm_wmma(const _Float16* __restrict__ A16, const _Float16* __restrict__ Bt,
               const float* __restrict__ bias, const float* __restrict__ residual,
               float* __restrict__ C, _Float16* __restrict__ C16,
               int M, int N, int K, int relu, int addpos) {
  __shared__ _Float16 As[2][BM * BKP];
  __shared__ _Float16 Bs[2][BN * BKP];

  const int tid  = threadIdx.x;
  const int wid  = tid >> 5;
  const int lane = tid & 31;
  const int wm   = wid & 1;
  const int wn   = wid >> 1;
  const int rowBase = blockIdx.y * BM;
  const int colBase = blockIdx.x * BN;
  const int nk = K / BK;

  const _Float16* Atile = A16 + (size_t)rowBase * K;
  const _Float16* Btile = Bt  + (size_t)colBase * K;

  v8f acc[4][2];
  v8f zero = {};
#pragma unroll
  for (int i = 0; i < 4; ++i)
#pragma unroll
    for (int j = 0; j < 2; ++j) acc[i][j] = zero;

#if __has_builtin(__builtin_amdgcn_tensor_load_to_lds)
  const unsigned ldsA0 = (unsigned)(uintptr_t)&As[0][0];
  const unsigned ldsA1 = (unsigned)(uintptr_t)&As[1][0];
  const unsigned ldsB0 = (unsigned)(uintptr_t)&Bs[0][0];
  const unsigned ldsB1 = (unsigned)(uintptr_t)&Bs[1][0];
  if (wid == 0) {  // wave 0 drives the Tensor Data Mover
    tdm_load_tile(ldsA0, Atile, K, M, BM);
    tdm_load_tile(ldsB0, Btile, K, N, BN);
  }
#endif

  for (int kt = 0; kt < nk; ++kt) {
    const int buf = kt & 1;
#if __has_builtin(__builtin_amdgcn_tensor_load_to_lds)
    if (wid == 0) {
      if (kt + 1 < nk) {
        const int k1 = (kt + 1) * BK;
        tdm_load_tile((kt & 1) ? ldsA0 : ldsA1, Atile + k1, K, M, BM);
        tdm_load_tile((kt & 1) ? ldsB0 : ldsB1, Btile + k1, K, N, BN);
        __builtin_amdgcn_s_wait_tensorcnt(2);   // current pair complete
      } else {
        __builtin_amdgcn_s_wait_tensorcnt(0);
      }
    }
#else
    {  // fallback: cooperative f16 copy (no conversion needed)
      const int k0 = kt * BK;
#pragma unroll
      for (int it = 0; it < 4; ++it) {
        int li = (it * 256 + tid) * 4;
        int r = li >> 5, c = li & 31;
        *reinterpret_cast<ulonglong1*>(&As[buf][r * BKP + c]) =
            *reinterpret_cast<const ulonglong1*>(Atile + (size_t)r * K + k0 + c);
        *reinterpret_cast<ulonglong1*>(&Bs[buf][r * BKP + c]) =
            *reinterpret_cast<const ulonglong1*>(Btile + (size_t)r * K + k0 + c);
      }
    }
#endif
    __syncthreads();   // buffer `buf` published to all waves

    v16h af[4], bf[2];
    const int kh = (lane >> 4) * 8;    // A frag: lanes 16-31 hold K+8
    const int kb = (lane >> 4) * 8;    // B frag: lanes 16-31 hold K+8
#pragma unroll
    for (int tm = 0; tm < 4; ++tm) {
      const _Float16* p = &As[buf][(wm * 64 + tm * 16 + (lane & 15)) * BKP];
#pragma unroll
      for (int e = 0; e < 16; ++e)
        af[tm][e] = p[((e >> 3) << 4) + kh + (e & 7)];
    }
#pragma unroll
    for (int tn = 0; tn < 2; ++tn) {
      const _Float16* p = &Bs[buf][(wn * 32 + tn * 16 + (lane & 15)) * BKP + kb];
#pragma unroll
      for (int e = 0; e < 16; ++e) bf[tn][e] = p[((e >> 3) << 4) + (e & 7)];
    }
#pragma unroll
    for (int tm = 0; tm < 4; ++tm)
#pragma unroll
      for (int tn = 0; tn < 2; ++tn)
        acc[tm][tn] = __builtin_amdgcn_wmma_f32_16x16x32_f16(
            false, af[tm], false, bf[tn], (short)0, acc[tm][tn], false, false);
    __syncthreads();   // all reads of `buf` done; TDM may overwrite it next round
  }

  // ---- epilogue: bias + residual + pos-enc + relu; f32 and/or f16 outputs
  const int rlo = (lane >> 4) << 3;
  const float pc = 9.2103403719761836f / 512.0f;  // ln(10000)/D
#pragma unroll
  for (int tm = 0; tm < 4; ++tm) {
#pragma unroll
    for (int tn = 0; tn < 2; ++tn) {
      const int col = colBase + wn * 32 + tn * 16 + (lane & 15);
      const float bv = bias[col];
#pragma unroll
      for (int r = 0; r < 8; ++r) {
        const int row = rowBase + wm * 64 + tm * 16 + rlo + r;
        float val = acc[tm][tn][r] + bv;
        if (residual) val += residual[(size_t)row * N + col];
        if (addpos) {
          const int s = row >> 5;  // row = s*BATCH + b, BATCH==32
          const float freq = __expf(-(float)(col & ~1) * pc);
          const float ang = (float)s * freq;
          val += (col & 1) ? __cosf(ang) : __sinf(ang);
        }
        if (relu) val = fmaxf(val, 0.0f);
        if (C)   C[(size_t)row * N + col] = val;
        if (C16) C16[(size_t)row * N + col] = (_Float16)val;
      }
    }
  }
}

// ---------------- sliding-window attention: block per (i,b), wave per head ------
__global__ __launch_bounds__(256)
void attn_kernel(const float* __restrict__ q, const float* __restrict__ k,
                 const float* __restrict__ v, _Float16* __restrict__ o16) {
  __shared__ float sQ[NH][HD];
  __shared__ float sP[NH][128];
  const int i = blockIdx.x, b = blockIdx.y;
  const int h = threadIdx.x >> 5, lane = threadIdx.x & 31;

  const float* qp = q + ((size_t)i * BATCH + b) * DM + h * HD;
  sQ[h][lane]      = qp[lane];
  sQ[h][lane + 32] = qp[lane + 32];
  __syncthreads();

  const int jlo = (i >= WIN - 1) ? (i - (WIN - 1)) : 0;
  const int cnt = i - jlo + 1;

  float s[4];
#pragma unroll
  for (int t = 0; t < 4; ++t) {
    const int j = jlo + t * 32 + lane;
    float sc = -1.0e30f;
    if (j <= i) {
      const float* kp = k + ((size_t)j * BATCH + b) * DM + h * HD;
      float d = 0.0f;
#pragma unroll 8
      for (int e = 0; e < HD; ++e) d += sQ[h][e] * kp[e];
      sc = d * 0.125f;
    }
    s[t] = sc;
  }
  float m = fmaxf(fmaxf(s[0], s[1]), fmaxf(s[2], s[3]));
#pragma unroll
  for (int off = 16; off > 0; off >>= 1) m = fmaxf(m, __shfl_xor(m, off, 32));

  float p[4], psum = 0.0f;
#pragma unroll
  for (int t = 0; t < 4; ++t) {
    p[t] = (s[t] > -1.0e29f) ? __expf(s[t] - m) : 0.0f;
    psum += p[t];
  }
#pragma unroll
  for (int off = 16; off > 0; off >>= 1) psum += __shfl_xor(psum, off, 32);
  const float inv = 1.0f / psum;
#pragma unroll
  for (int t = 0; t < 4; ++t) sP[h][t * 32 + lane] = p[t] * inv;
  __syncthreads();

  float a0 = 0.0f, a1 = 0.0f;
  for (int t = 0; t < cnt; ++t) {
    const float pj = sP[h][t];
    const float* vp = v + ((size_t)(jlo + t) * BATCH + b) * DM + h * HD;
    a0 += pj * vp[lane];
    a1 += pj * vp[lane + 32];
  }
  _Float16* op = o16 + ((size_t)i * BATCH + b) * DM + h * HD;
  op[lane]      = (_Float16)a0;
  op[lane + 32] = (_Float16)a1;
}

// ---------------- layernorm: block per row of D=512, f32 + optional f16 out ------
__global__ __launch_bounds__(256)
void ln_kernel(const float* __restrict__ y, const float* __restrict__ sc,
               const float* __restrict__ bi, float* __restrict__ out,
               _Float16* __restrict__ out16) {
  __shared__ float red[16];
  const int row = blockIdx.x, t = threadIdx.x;
  const float* yp = y + (size_t)row * DM;
  const float x0 = yp[t], x1 = yp[t + 256];
  float s = x0 + x1, s2 = x0 * x0 + x1 * x1;
#pragma unroll
  for (int off = 16; off > 0; off >>= 1) {
    s  += __shfl_xor(s, off, 32);
    s2 += __shfl_xor(s2, off, 32);
  }
  const int wid = t >> 5, lane = t & 31;
  if (lane == 0) { red[wid] = s; red[8 + wid] = s2; }
  __syncthreads();
  if (t == 0) {
    float S = 0.0f, S2 = 0.0f;
#pragma unroll
    for (int w = 0; w < 8; ++w) { S += red[w]; S2 += red[8 + w]; }
    red[0] = S; red[1] = S2;
  }
  __syncthreads();
  const float mean = red[0] * (1.0f / DM);
  const float var  = red[1] * (1.0f / DM) - mean * mean;
  const float r    = rsqrtf(var + 1e-5f);
  const float v0 = (x0 - mean) * r * sc[t] + bi[t];
  const float v1 = (x1 - mean) * r * sc[t + 256] + bi[t + 256];
  out[(size_t)row * DM + t]       = v0;
  out[(size_t)row * DM + t + 256] = v1;
  if (out16) {
    out16[(size_t)row * DM + t]       = (_Float16)v0;
    out16[(size_t)row * DM + t + 256] = (_Float16)v1;
  }
}

// ---------------- one-time converters ----------------
__global__ void convert_f16(const float* __restrict__ in, _Float16* __restrict__ out, int n) {
  int i = blockIdx.x * 256 + threadIdx.x;
  if (i < n) out[i] = (_Float16)in[i];
}
// W [K,N] f32 -> Wt [N,K] f16
__global__ void convert_transpose_f16(const float* __restrict__ in, _Float16* __restrict__ out,
                                      int K, int N) {
  int i = blockIdx.x * 256 + threadIdx.x;
  if (i < K * N) {
    int k = i / N, n = i % N;
    out[(size_t)n * K + k] = (_Float16)in[i];
  }
}

// ---------------- driver ----------------
extern "C" void kernel_launch(void* const* d_in, const int* in_sizes, int n_in,
                              void* d_out, int out_size, void* d_ws, size_t ws_size,
                              hipStream_t stream) {
  const float* x     = (const float*)d_in[0];
  const float* W_emb = (const float*)d_in[1];
  const float* b_emb = (const float*)d_in[2];
  const float* Wq    = (const float*)d_in[3];
  const float* bq    = (const float*)d_in[4];
  const float* Wk    = (const float*)d_in[5];
  const float* bk    = (const float*)d_in[6];
  const float* Wv    = (const float*)d_in[7];
  const float* bv    = (const float*)d_in[8];
  const float* Wo    = (const float*)d_in[9];
  const float* bo    = (const float*)d_in[10];
  const float* ln1_s = (const float*)d_in[11];
  const float* ln1_b = (const float*)d_in[12];
  const float* W1    = (const float*)d_in[13];
  const float* b1    = (const float*)d_in[14];
  const float* W2    = (const float*)d_in[15];
  const float* b2    = (const float*)d_in[16];
  const float* ln2_s = (const float*)d_in[17];
  const float* ln2_b = (const float*)d_in[18];
  float* out = (float*)d_out;

  char* ws = (char*)d_ws;
  const size_t HB  = (size_t)MROWS * DM * sizeof(float);      // 32.768 MB
  const size_t HB16 = HB / 2;
  float*    h    = (float*)(ws);
  float*    qb   = (float*)(ws + 1 * HB);
  float*    kb_  = (float*)(ws + 2 * HB);
  float*    vb   = (float*)(ws + 3 * HB);
  float*    tb   = qb;  // reuse
  _Float16* h16  = (_Float16*)(ws + 4 * HB);
  _Float16* o16  = (_Float16*)(ws + 4 * HB + HB16);
  _Float16* ff16 = (_Float16*)(ws + 4 * HB + 2 * HB16);        // MROWS x FFD f16
  _Float16* x16  = (_Float16*)(ws + 4 * HB + 2 * HB16 + (size_t)MROWS * FFD * 2);
  _Float16* wts  = (_Float16*)((char*)x16 + (size_t)MROWS * 256 * 2);

  _Float16* embT = wts;                                        // [512,256]
  const size_t embTsz = (size_t)DM * 256;
  const size_t projSz = (size_t)DM * DM;                       // 262144
  const size_t ffSz   = (size_t)DM * FFD;                      // 1048576
  const size_t layerSz = 4 * projSz + 2 * ffSz;
  _Float16* lw = embT + embTsz;

  const dim3 blk(256);
  const dim3 gD(DM / BN, MROWS / BM);    // 4  x 125
  const dim3 gF(FFD / BN, MROWS / BM);   // 16 x 125

  // ---- one-time conversions (f16, weights pre-transposed to [N,K])
  convert_f16<<<(MROWS * 256 + 255) / 256, blk, 0, stream>>>(x, x16, MROWS * 256);
  convert_transpose_f16<<<(256 * DM + 255) / 256, blk, 0, stream>>>(W_emb, embT, 256, DM);
  for (int l = 0; l < NL; ++l) {
    _Float16* base = lw + (size_t)l * layerSz;
    const int nPP = (DM * DM + 255) / 256, nFF = (DM * FFD + 255) / 256;
    convert_transpose_f16<<<nPP, blk, 0, stream>>>(Wq + (size_t)l * DM * DM, base + 0 * projSz, DM, DM);
    convert_transpose_f16<<<nPP, blk, 0, stream>>>(Wk + (size_t)l * DM * DM, base + 1 * projSz, DM, DM);
    convert_transpose_f16<<<nPP, blk, 0, stream>>>(Wv + (size_t)l * DM * DM, base + 2 * projSz, DM, DM);
    convert_transpose_f16<<<nPP, blk, 0, stream>>>(Wo + (size_t)l * DM * DM, base + 3 * projSz, DM, DM);
    convert_transpose_f16<<<nFF, blk, 0, stream>>>(W1 + (size_t)l * DM * FFD, base + 4 * projSz, DM, FFD);
    convert_transpose_f16<<<nFF, blk, 0, stream>>>(W2 + (size_t)l * FFD * DM, base + 4 * projSz + ffSz, FFD, DM);
  }

  // ---- embedding + positional encoding -> h (f32) + h16
  gemm_wmma<<<gD, blk, 0, stream>>>(x16, embT, b_emb, nullptr, h, h16,
                                    MROWS, DM, 256, 0, 1);

  for (int l = 0; l < NL; ++l) {
    _Float16* base = lw + (size_t)l * layerSz;
    _Float16* WqT = base + 0 * projSz;
    _Float16* WkT = base + 1 * projSz;
    _Float16* WvT = base + 2 * projSz;
    _Float16* WoT = base + 3 * projSz;
    _Float16* W1T = base + 4 * projSz;
    _Float16* W2T = base + 4 * projSz + ffSz;

    gemm_wmma<<<gD, blk, 0, stream>>>(h16, WqT, bq + l * DM, nullptr, qb, nullptr,
                                      MROWS, DM, DM, 0, 0);
    gemm_wmma<<<gD, blk, 0, stream>>>(h16, WkT, bk + l * DM, nullptr, kb_, nullptr,
                                      MROWS, DM, DM, 0, 0);
    gemm_wmma<<<gD, blk, 0, stream>>>(h16, WvT, bv + l * DM, nullptr, vb, nullptr,
                                      MROWS, DM, DM, 0, 0);
    attn_kernel<<<dim3(SEQ, BATCH), blk, 0, stream>>>(qb, kb_, vb, o16);
    gemm_wmma<<<gD, blk, 0, stream>>>(o16, WoT, bo + l * DM, h, tb, nullptr,
                                      MROWS, DM, DM, 0, 0);
    ln_kernel<<<MROWS, blk, 0, stream>>>(tb, ln1_s + l * DM, ln1_b + l * DM, h, h16);
    gemm_wmma<<<gF, blk, 0, stream>>>(h16, W1T, b1 + l * FFD, nullptr, nullptr, ff16,
                                      MROWS, FFD, DM, 1, 0);
    gemm_wmma<<<gD, blk, 0, stream>>>(ff16, W2T, b2 + l * DM, h, tb, nullptr,
                                      MROWS, DM, FFD, 0, 0);
    if (l == NL - 1) {
      ln_kernel<<<MROWS, blk, 0, stream>>>(tb, ln2_s + l * DM, ln2_b + l * DM, out, nullptr);
    } else {
      ln_kernel<<<MROWS, blk, 0, stream>>>(tb, ln2_s + l * DM, ln2_b + l * DM, h, h16);
    }
  }
}
